// EdgeGatedGraphConv_52106543235177
// MI455X (gfx1250) — compile-verified, matrix operand order checked by reading, not run on hardware
//
#include <hip/hip_runtime.h>
#include <hip/hip_bf16.h>
#include <math.h>

// ---------------------------------------------------------------------------
// EdgeGatedGraphConv on gfx1250 (MI455X): bf16 WMMA GEMMs, TDM tile loads,
// fused gating/LN epilogues. N=10000 nodes, E=320000 edges, D=T=128.
// ---------------------------------------------------------------------------

typedef __attribute__((ext_vector_type(16))) __bf16   v16bf;
typedef __attribute__((ext_vector_type(8)))  float    v8f;
typedef __attribute__((ext_vector_type(4)))  unsigned u32x4;
typedef __attribute__((ext_vector_type(8)))  int      i32x8;

#define DD 128

// --------------------------- TDM tile load ---------------------------------
// 2D tile load via Tensor Data Mover: tile = tileRows x 128 f32, row stride
// 128 elements. rowsAvail = rows remaining in the tensor (OOB rows -> zero).
// D# group0/group1 packed per CDNA5 ISA ch.8; groups 2/3 NULL (2D tensor).
__device__ __forceinline__ void tdm_load_tile_2d(const float* __restrict__ gsrc,
                                                 unsigned ldsOff,
                                                 unsigned rowsAvail,
                                                 unsigned tileRows) {
  unsigned long long ga = (unsigned long long)gsrc;
  u32x4 g0;
  g0.x = 1u;                                            // count=1, user mode
  g0.y = ldsOff;                                        // lds_addr (bytes)
  g0.z = (unsigned)ga;                                  // global_addr[31:0]
  g0.w = (unsigned)((ga >> 32) & 0x01FFFFFFull)         // global_addr[56:32]
         | (2u << 30);                                  // type=2 ("image")
  i32x8 g1;
  g1[0] = 0x20000;                                      // data_size=2 (4 bytes)
  g1[1] = (int)(128u << 16);                            // tensor_dim0 = 128
  g1[2] = (int)((rowsAvail & 0xFFFFu) << 16);           // tensor_dim1[15:0]
  g1[3] = (int)(((rowsAvail >> 16) & 0xFFFFu)           // tensor_dim1[31:16]
                | (128u << 16));                        // tile_dim0 = 128
  g1[4] = (int)(tileRows & 0xFFFFu);                    // tile_dim1
  g1[5] = 128;                                          // tensor_dim0_stride
  g1[6] = 0;
  g1[7] = 0;
  asm volatile("tensor_load_to_lds %0, %1" :: "s"(g0), "s"(g1) : "memory");
}

__device__ __forceinline__ unsigned lds_off_of(const void* p) {
  // generic LDS pointer = {shared aperture, 32-bit LDS offset}
  return (unsigned)(unsigned long long)p;
}

// --------------------------- WMMA fragments --------------------------------
// bf16 A/B 16x32 fragment layout: VGPR j holds K pair {2*(j&3) + (j>>2)*16},
// +8 for lanes 16..31. Pairs contiguous in K.
__device__ __forceinline__ v16bf frag_ld(const __bf16* __restrict__ row,
                                         int kbase, int half) {
  union { v16bf v; unsigned u[8]; } f;
  const unsigned* p = reinterpret_cast<const unsigned*>(row + kbase + half * 8);
#pragma unroll
  for (int j = 0; j < 8; ++j) {
    const int koff = 2 * (j & 3) + ((j >> 2) << 4);     // 0,2,4,6,16,18,20,22
    f.u[j] = p[koff >> 1];
  }
  return f.v;
}

// Same fragment, sourced from an f32 LDS row (convert at build time).
__device__ __forceinline__ v16bf frag_ld_f32(const float* __restrict__ row,
                                             int kbase, int half) {
  union { v16bf v; __bf16 h[16]; } f;
  const float* p = row + kbase + half * 8;
#pragma unroll
  for (int j = 0; j < 8; ++j) {
    const int koff = 2 * (j & 3) + ((j >> 2) << 4);
    f.h[2 * j]     = (__bf16)p[koff];
    f.h[2 * j + 1] = (__bf16)p[koff + 1];
  }
  return f.v;
}

// One wave: 16x128 output tile, K=128 (4 steps of 32). A rows from f32 LDS,
// B from bf16 transposed-weight LDS (WshT[n*128+k] = W[k][n]).
__device__ __forceinline__ void gemm_tile(const float* __restrict__ Arow,
                                          const __bf16* __restrict__ WshT,
                                          int lane, v8f acc[8]) {
  const int half = lane >> 4, cl = lane & 15;
  v16bf a[4];
#pragma unroll
  for (int kk = 0; kk < 4; ++kk) a[kk] = frag_ld_f32(Arow, kk * 32, half);
#pragma unroll
  for (int nt = 0; nt < 8; ++nt) {
    const __bf16* brow = WshT + (nt * 16 + cl) * 128;
#pragma unroll
    for (int kk = 0; kk < 4; ++kk) {
      v16bf b = frag_ld(brow, kk * 32, half);
      acc[nt] = __builtin_amdgcn_wmma_f32_16x16x32_bf16(
          false, a[kk], false, b, (short)0, acc[nt], false, false);
    }
  }
}

__device__ __forceinline__ float fsigmoid(float x) {
  return 1.0f / (1.0f + __expf(-x));
}

// ---------------------------------------------------------------------------
// Kernel 1: node-side GEMMs. Block = 128 threads (4 waves) -> 64 node rows.
// LDS: 32KB f32 A tile (TDM-loaded) + 32KB bf16 transposed weight.
//   e_dst = node@Wdg + bdg ; Bh = node@Wdu + bdu ; xs = node@Wsu + bsu
//   e_src = node@Wsg + b_sg + time@Wtp + b_tp
// ---------------------------------------------------------------------------
__global__ __launch_bounds__(128) void node_gemm_k(
    const float* __restrict__ node, const float* __restrict__ timef,
    const float* __restrict__ Wtp, const float* __restrict__ btp,
    const float* __restrict__ Wsg, const float* __restrict__ bsg,
    const float* __restrict__ Wdg, const float* __restrict__ bdg,
    const float* __restrict__ Wsu, const float* __restrict__ bsu,
    const float* __restrict__ Wdu, const float* __restrict__ bdu,
    float* __restrict__ esrc, float* __restrict__ edstv,
    float* __restrict__ Bh, float* __restrict__ xs, int nN) {
  __shared__ float  Af32[64 * 128];     // 32 KB
  __shared__ __bf16 WshT[128 * 128];    // 32 KB
  const int t = threadIdx.x, lane = t & 31, wave = t >> 5;
  const int half = lane >> 4, cl = lane & 15;
  const int rowBase = blockIdx.x * 64;
  const unsigned aOff = lds_off_of(&Af32[0]);

  auto load_W = [&](const float* __restrict__ W) {
#pragma unroll 4
    for (int i = 0; i < 128; ++i) {
      int idx = t + i * 128;
      WshT[(idx & 127) * 128 + (idx >> 7)] = (__bf16)W[idx];
    }
  };
  auto init_b = [&](v8f* acc, const float* b1, const float* b2) {
#pragma unroll
    for (int nt = 0; nt < 8; ++nt) {
      float bv = b1[nt * 16 + cl] + (b2 ? b2[nt * 16 + cl] : 0.0f);
#pragma unroll
      for (int r = 0; r < 8; ++r) acc[nt][r] = bv;
    }
  };
  auto store = [&](float* __restrict__ dst, v8f* acc) {
#pragma unroll
    for (int nt = 0; nt < 8; ++nt) {
      int c = nt * 16 + cl;
#pragma unroll
      for (int r = 0; r < 8; ++r) {
        int grow = rowBase + wave * 16 + half * 8 + r;
        if (grow < nN) dst[(size_t)grow * DD + c] = acc[nt][r];
      }
    }
  };

  const float* arow = Af32 + (wave * 16 + cl) * 128;
  v8f acc[8];

  // TDM: node tile (OOB rows beyond nN auto-zero via tensor_dim1)
  if (wave == 0) {
    tdm_load_tile_2d(node + (size_t)rowBase * DD, aOff,
                     (unsigned)(nN - rowBase), 64u);
    __builtin_amdgcn_s_wait_tensorcnt(0);
  }
  load_W(Wdg);
  __syncthreads();

  init_b(acc, bdg, nullptr);
  gemm_tile(arow, WshT, lane, acc);
  store(edstv, acc);
  __syncthreads();

  load_W(Wdu);
  __syncthreads();
  init_b(acc, bdu, nullptr);
  gemm_tile(arow, WshT, lane, acc);
  store(Bh, acc);
  __syncthreads();

  load_W(Wsu);
  __syncthreads();
  init_b(acc, bsu, nullptr);
  gemm_tile(arow, WshT, lane, acc);
  store(xs, acc);
  __syncthreads();

  load_W(Wsg);
  __syncthreads();
  init_b(acc, bsg, btp);
  gemm_tile(arow, WshT, lane, acc);   // partial e_src, keep acc live
  __syncthreads();

  // Swap A tile: time features; accumulate time@Wtp into e_src
  if (wave == 0) {
    tdm_load_tile_2d(timef + (size_t)rowBase * DD, aOff,
                     (unsigned)(nN - rowBase), 64u);
    __builtin_amdgcn_s_wait_tensorcnt(0);
  }
  load_W(Wtp);
  __syncthreads();
  gemm_tile(arow, WshT, lane, acc);
  store(esrc, acc);
}

// ---------------------------------------------------------------------------
// Kernel 2: edge GEMM + gates + edge LN/silu output + scatter-add atomics.
// Block = 128 threads (4 waves) -> 64 edges. E % 64 == 0.
// ---------------------------------------------------------------------------
__global__ __launch_bounds__(128) void edge_k(
    const float* __restrict__ ef, const int* __restrict__ src,
    const int* __restrict__ dst, const float* __restrict__ Weg,
    const float* __restrict__ beg, const float* __restrict__ esrc,
    const float* __restrict__ edstv, const float* __restrict__ Bh,
    const float* __restrict__ lng, const float* __restrict__ lnb,
    float* __restrict__ ssh, float* __restrict__ ssum,
    float* __restrict__ yout, int nE) {
  __shared__ float  Af32[64 * 128];     // 32 KB (also serves the residual)
  __shared__ __bf16 WshT[128 * 128];    // 32 KB
  const int t = threadIdx.x, lane = t & 31, wave = t >> 5;
  const int half = lane >> 4, cl = lane & 15;
  const long eb = (long)blockIdx.x * 64;

  // TDM: edge-feature tile -> LDS (f32)
  if (wave == 0) {
    tdm_load_tile_2d(ef + eb * DD, lds_off_of(&Af32[0]),
                     (unsigned)(nE - eb), 64u);
    __builtin_amdgcn_s_wait_tensorcnt(0);
  }
#pragma unroll 4
  for (int i = 0; i < 128; ++i) {
    int idx = t + i * 128;
    WshT[(idx & 127) * 128 + (idx >> 7)] = (__bf16)Weg[idx];
  }
  __syncthreads();

  v8f acc[8];
#pragma unroll
  for (int nt = 0; nt < 8; ++nt) {
    float bv = beg[nt * 16 + cl];
#pragma unroll
    for (int r = 0; r < 8; ++r) acc[nt][r] = bv;
  }
  gemm_tile(Af32 + (wave * 16 + cl) * 128, WshT, lane, acc);

  // this lane's 8 edge rows: e0 + r (r = 0..7)
  const long e0 = eb + wave * 16 + half * 8;
  const int rloc0 = wave * 16 + half * 8;
  int sidx[8], didx[8];
#pragma unroll
  for (int r = 0; r < 8; ++r) { sidx[r] = src[e0 + r]; didx[r] = dst[e0 + r]; }

  // m += e_src[src] + e_dst[dst]  (L2-resident gathers, 64B-coalesced per half)
#pragma unroll
  for (int nt = 0; nt < 8; ++nt) {
    int c = nt * 16 + cl;
#pragma unroll
    for (int r = 0; r < 8; ++r)
      acc[nt][r] += esrc[(size_t)sidx[r] * DD + c] +
                    edstv[(size_t)didx[r] * DD + c];
  }

  // LayerNorm stats per edge row: reduce 8 regs x 16 lanes (half-wave shfl).
  float s[8], sq[8];
#pragma unroll
  for (int r = 0; r < 8; ++r) {
    s[r] = 0.0f; sq[r] = 0.0f;
#pragma unroll
    for (int nt = 0; nt < 8; ++nt) {
      float v = acc[nt][r];
      s[r] += v; sq[r] += v * v;
    }
  }
#pragma unroll
  for (int m = 1; m < 16; m <<= 1) {
#pragma unroll
    for (int r = 0; r < 8; ++r) {
      s[r]  += __shfl_xor(s[r],  m, 32);
      sq[r] += __shfl_xor(sq[r], m, 32);
    }
  }
  float mu[8], rstd[8];
#pragma unroll
  for (int r = 0; r < 8; ++r) {
    mu[r] = s[r] * (1.0f / 128.0f);
    float var = sq[r] * (1.0f / 128.0f) - mu[r] * mu[r];
    rstd[r] = rsqrtf(var + 1e-5f);
  }

  // y = edge_feats + silu(LN(m));  scatter-add sum_sigma / sum_sigma_h
#pragma unroll
  for (int nt = 0; nt < 8; ++nt) {
    int c = nt * 16 + cl;
    float g = lng[c], b = lnb[c];
#pragma unroll
    for (int r = 0; r < 8; ++r) {
      long e = e0 + r;
      float v = acc[nt][r];
      float sig = fsigmoid(v);
      float ln = (v - mu[r]) * rstd[r] * g + b;
      yout[e * DD + c] = Af32[(rloc0 + r) * 128 + c] + ln * fsigmoid(ln);
      atomicAdd(&ssum[(size_t)didx[r] * DD + c], sig);
      atomicAdd(&ssh[(size_t)didx[r] * DD + c],
                Bh[(size_t)sidx[r] * DD + c] * sig);
    }
  }
}

// ---------------------------------------------------------------------------
// Kernel 3: node epilogue.  x = node + silu(LN(xs + ssh/(ss+1e-6)))
// ---------------------------------------------------------------------------
__global__ __launch_bounds__(128) void node_final_k(
    const float* __restrict__ node, const float* __restrict__ xsv,
    const float* __restrict__ ssh, const float* __restrict__ ssum,
    const float* __restrict__ g, const float* __restrict__ b,
    float* __restrict__ xout) {
  __shared__ float redS[4], redQ[4];
  const int c = threadIdx.x;
  const size_t nid = blockIdx.x;
  float base = xsv[nid * DD + c] +
               ssh[nid * DD + c] / (ssum[nid * DD + c] + 1e-6f);
  float sv = base, sq = base * base;
#pragma unroll
  for (int m = 1; m < 32; m <<= 1) {
    sv += __shfl_xor(sv, m, 32);
    sq += __shfl_xor(sq, m, 32);
  }
  if ((c & 31) == 0) { redS[c >> 5] = sv; redQ[c >> 5] = sq; }
  __syncthreads();
  float tot  = redS[0] + redS[1] + redS[2] + redS[3];
  float totq = redQ[0] + redQ[1] + redQ[2] + redQ[3];
  float mu = tot * (1.0f / 128.0f);
  float var = totq * (1.0f / 128.0f) - mu * mu;
  float ln = (base - mu) * rsqrtf(var + 1e-5f) * g[c] + b[c];
  xout[nid * DD + c] = node[nid * DD + c] + ln * fsigmoid(ln);
}

__global__ void zero_k(float* __restrict__ p, long n) {
  long i = (long)blockIdx.x * 256 + threadIdx.x;
  if (i < n) p[i] = 0.0f;
}

// ---------------------------------------------------------------------------
extern "C" void kernel_launch(void* const* d_in, const int* in_sizes, int n_in,
                              void* d_out, int out_size, void* d_ws,
                              size_t ws_size, hipStream_t stream) {
  (void)n_in; (void)out_size; (void)ws_size;
  const float* node  = (const float*)d_in[0];
  const float* ef    = (const float*)d_in[1];
  const float* timef = (const float*)d_in[2];
  const int*   src   = (const int*)d_in[3];
  const int*   dst   = (const int*)d_in[4];
  const float* Wtp = (const float*)d_in[5];  const float* btp = (const float*)d_in[6];
  const float* Wsg = (const float*)d_in[7];  const float* bsg = (const float*)d_in[8];
  const float* Wdg = (const float*)d_in[9];  const float* bdg = (const float*)d_in[10];
  const float* Weg = (const float*)d_in[11]; const float* beg = (const float*)d_in[12];
  const float* Wsu = (const float*)d_in[13]; const float* bsu = (const float*)d_in[14];
  const float* Wdu = (const float*)d_in[15]; const float* bdu = (const float*)d_in[16];
  const float* lneg = (const float*)d_in[17]; const float* lneb = (const float*)d_in[18];
  const float* lnng = (const float*)d_in[19]; const float* lnnb = (const float*)d_in[20];

  const int N = in_sizes[0] / DD;   // 10000
  const int E = in_sizes[3];        // 320000
  const size_t ND = (size_t)N * DD;

  float* ws    = (float*)d_ws;
  float* esrc  = ws;
  float* edstv = ws + 1 * ND;
  float* Bh    = ws + 2 * ND;
  float* xs    = ws + 3 * ND;
  float* ssh   = ws + 4 * ND;
  float* ssum  = ws + 5 * ND;

  float* xout = (float*)d_out;
  float* yout = xout + ND;

  const long zn = (long)2 * ND;  // ssh + ssum are contiguous
  zero_k<<<(int)((zn + 255) / 256), 256, 0, stream>>>(ssh, zn);

  node_gemm_k<<<(N + 63) / 64, 128, 0, stream>>>(
      node, timef, Wtp, btp, Wsg, bsg, Wdg, bdg, Wsu, bsu, Wdu, bdu,
      esrc, edstv, Bh, xs, N);

  edge_k<<<E / 64, 128, 0, stream>>>(
      ef, src, dst, Weg, beg, esrc, edstv, Bh, lneg, lneb, ssh, ssum, yout, E);

  node_final_k<<<N, 128, 0, stream>>>(node, xs, ssh, ssum, lnng, lnnb, xout);
}
